// Model_78176994722128
// MI455X (gfx1250) — compile-verified, matrix operand order checked by reading, not run on hardware
//
#include <hip/hip_runtime.h>
#include <hip/hip_bf16.h>
#include <math.h>
#include <cstdint>

// ---------------------------------------------------------------------------
// MI455X / gfx1250: bf16 WMMA for all GEMM-shaped math, fp32 LN/softmax/
// residual/GELU. Compute-bound (~200 GFLOP vs ~60MB resident => L2-held).
// This revision: (a) async global->LDS tile staging (ASYNCcnt) with double
// buffering so DMA of tile k+1 overlaps WMMA of tile k; (b) all transposes
// hoisted out of inner loops (weights stored N-major at convert time, V
// stored channel-major), so every inner-loop LDS op is an aligned b128.
// ---------------------------------------------------------------------------

typedef __attribute__((ext_vector_type(16))) __bf16 bf16x16;
typedef __attribute__((ext_vector_type(8)))  float  f32x8;

union Frag { uint4 u[2]; bf16x16 v; };

static constexpr int LAY = 4, Bsz = 2, Tsz = 2048, Ssz = 512, Wd = 512, Hh = 8, CHd = 64;
static constexpr int BT = Bsz * Tsz;

__device__ __forceinline__ unsigned short f2bf(float f) {
  unsigned u = __float_as_uint(f);
  unsigned r = 0x7FFFu + ((u >> 16) & 1u);   // round-to-nearest-even
  return (unsigned short)((u + r) >> 16);
}

__device__ __forceinline__ f32x8 zero8() {
  f32x8 z;
#pragma unroll
  for (int i = 0; i < 8; i++) z[i] = 0.f;
  return z;
}

// --- async 16B global->LDS copy (GLOBAL_LOAD_ASYNC_TO_LDS_B128, ASYNCcnt) ---
// LDS operand is the wave-relative LDS byte offset = low 32 bits of the flat
// pointer to a __shared__ object (LDS aperture lives in the high bits).
__device__ __forceinline__ void stage16(void* lds, const void* g) {
  unsigned lo = (unsigned)(uintptr_t)lds;
  asm volatile("global_load_async_to_lds_b128 %0, %1, off"
               :: "v"(lo), "v"(g) : "memory");
}
__device__ __forceinline__ void wait_async0() {
  asm volatile("s_wait_asynccnt 0x0" ::: "memory");
}

// --------------------------- fp32 -> bf16 convert ---------------------------
__global__ void k_cvt(const float* __restrict__ in, unsigned short* __restrict__ out, int n) {
  for (int i = blockIdx.x * blockDim.x + threadIdx.x; i < n; i += gridDim.x * blockDim.x)
    out[i] = f2bf(in[i]);
}

// convert + transpose weights: in (nmat, K, N) fp32 -> out (nmat, N, K) bf16
__global__ void k_cvt_wt(const float* __restrict__ in, unsigned short* __restrict__ out,
                         int K, int N, int total) {
  int kn = K * N;
  for (int i = blockIdx.x * blockDim.x + threadIdx.x; i < total; i += gridDim.x * blockDim.x) {
    int mat = i / kn, rem = i - mat * kn;
    int k = rem / N, n = rem - k * N;
    out[(size_t)mat * kn + (size_t)n * K + k] = f2bf(in[i]);
  }
}

// convert + transpose cond_v: in (B,S,H*CH) fp32 -> out (B,H,CH,S) bf16
__global__ void k_cvt_vt(const float* __restrict__ in, unsigned short* __restrict__ out, int total) {
  for (int i = blockIdx.x * blockDim.x + threadIdx.x; i < total; i += gridDim.x * blockDim.x) {
    int b = i / (Ssz * Wd), rem = i - b * (Ssz * Wd);
    int s = rem / Wd, j = rem - s * Wd;
    int h = j >> 6, c = j & 63;
    out[(((size_t)(b * Hh + h)) * CHd + c) * Ssz + s] = f2bf(in[i]);
  }
}

// ------------------------------- LayerNorm ----------------------------------
__global__ __launch_bounds__(256) void k_ln(const float* __restrict__ x,
                                            const float* __restrict__ g,
                                            const float* __restrict__ b,
                                            unsigned short* __restrict__ h) {
  int row = blockIdx.x, t = threadIdx.x;
  const float* xr = x + (size_t)row * Wd;
  float v0 = xr[t], v1 = xr[t + 256];
  float s = v0 + v1, q = v0 * v0 + v1 * v1;
#pragma unroll
  for (int m = 16; m >= 1; m >>= 1) { s += __shfl_xor(s, m, 32); q += __shfl_xor(q, m, 32); }
  __shared__ float rs[8], rq[8];
  if ((t & 31) == 0) { rs[t >> 5] = s; rq[t >> 5] = q; }
  __syncthreads();
  s = 0.f; q = 0.f;
#pragma unroll
  for (int i = 0; i < 8; i++) { s += rs[i]; q += rq[i]; }
  float mean = s * (1.f / Wd);
  float var  = q * (1.f / Wd) - mean * mean;
  float rstd = rsqrtf(var + 1e-5f);
  unsigned short* hr = h + (size_t)row * Wd;
  hr[t]       = f2bf((v0 - mean) * rstd * g[t]       + b[t]);
  hr[t + 256] = f2bf((v1 - mean) * rstd * g[t + 256] + b[t + 256]);
}

// --------------------------------- GEMM -------------------------------------
// C[M,N] = A[M,K](bf16, row-major) @ Wt[N,K](bf16, N-major!) + bias
//   MODE 1: f32 out = resid + (..)            (residual projections, in-place x)
//   MODE 2: bf16 out = gelu(..)               (MLP fc)
//   MODE 3: qkv epilogue: q/k parts row-major to outp, v part transposed
//           to vt_out as (B,H,CH,T) for async V staging in attention
// Block tile 128x128, 8 waves (4x2), wave 32x64 (2x4 accs), K-step 32,
// double-buffered async LDS staging.
template <int MODE>
__global__ __launch_bounds__(256) void k_gemm(const unsigned short* __restrict__ A, int lda,
                                              const unsigned short* __restrict__ Bw, int ldb,
                                              const float* __restrict__ bias,
                                              const float* __restrict__ resid,
                                              void* __restrict__ outp,
                                              unsigned short* __restrict__ vt_out,
                                              int M, int N, int K) {
  __shared__ unsigned short Al[2][128 * 48];   // [buf][row][k] pitch 48 (96B rows)
  __shared__ unsigned short Bl[2][128 * 48];   // [buf][n]  [k] pitch 48
  int t = threadIdx.x;
  int lane = t & 31, wv = t >> 5;
  int l16 = lane & 15, hi = lane >> 4;
  int wm = wv >> 1, wn = wv & 1;
  int bM = blockIdx.y * 128, bN = blockIdx.x * 128;

  f32x8 acc[2][4];
#pragma unroll
  for (int mi = 0; mi < 2; mi++)
#pragma unroll
    for (int ni = 0; ni < 4; ni++) acc[mi][ni] = zero8();

  auto stageAB = [&](int kb, int buf) {
    int idx = t;
#pragma unroll
    for (int pp = 0; pp < 2; pp++, idx += 256) {
      int r = idx >> 2, c = (idx & 3) * 8;
      stage16(&Al[buf][r * 48 + c], A + (size_t)(bM + r) * lda + kb + c);
      stage16(&Bl[buf][r * 48 + c], Bw + (size_t)(bN + r) * ldb + kb + c);
    }
  };

  stageAB(0, 0);
  for (int kb = 0; kb < K; kb += 32) {
    int buf = (kb >> 5) & 1;
    wait_async0();
    __syncthreads();
    if (kb + 32 < K) stageAB(kb + 32, buf ^ 1);  // DMA overlaps compute below

    Frag af[2], bfr[4];
#pragma unroll
    for (int mi = 0; mi < 2; mi++) {
      const unsigned short* p = &Al[buf][(wm * 32 + mi * 16 + l16) * 48];
      af[mi].u[0] = *(const uint4*)(p + hi * 8);
      af[mi].u[1] = *(const uint4*)(p + hi * 8 + 16);
    }
#pragma unroll
    for (int ni = 0; ni < 4; ni++) {
      const unsigned short* p = &Bl[buf][(wn * 64 + ni * 16 + l16) * 48];
      bfr[ni].u[0] = *(const uint4*)(p + hi * 8);
      bfr[ni].u[1] = *(const uint4*)(p + hi * 8 + 16);
    }
#pragma unroll
    for (int mi = 0; mi < 2; mi++)
#pragma unroll
      for (int ni = 0; ni < 4; ni++)
        acc[mi][ni] = __builtin_amdgcn_wmma_f32_16x16x32_bf16(
            false, af[mi].v, false, bfr[ni].v, (short)0, acc[mi][ni], false, false);
  }

#pragma unroll
  for (int mi = 0; mi < 2; mi++) {
#pragma unroll
    for (int ni = 0; ni < 4; ni++) {
      int col = bN + wn * 64 + ni * 16 + l16;
      float bv = bias[col];
#pragma unroll
      for (int r = 0; r < 8; r++) {
        int row = bM + wm * 32 + mi * 16 + r + hi * 8;
        float v = acc[mi][ni][r] + bv;
        size_t off = (size_t)row * N + col;
        if (MODE == 1) {
          ((float*)outp)[off] = v + resid[off];
        } else if (MODE == 2) {
          float gl = 0.5f * v * (1.f + erff(v * 0.70710678118f));  // exact GELU
          ((unsigned short*)outp)[off] = f2bf(gl);
        } else {  // MODE 3: qkv with transposed V
          int head = col / (3 * CHd);
          int j = col - head * 3 * CHd;
          if (j < 2 * CHd) {
            ((unsigned short*)outp)[off] = f2bf(v);
          } else {
            int c = j - 2 * CHd;
            int bb = row >> 11, tt = row & (Tsz - 1);
            vt_out[(((size_t)(bb * Hh + head)) * CHd + c) * Tsz + tt] = f2bf(v);
          }
        }
      }
    }
  }
}

// ------------------------------ Flash attention ------------------------------
// grid (B*H, T/128), 8 waves; wave owns 16 query rows (Q frags pinned in regs).
// K row-major [s][ch], V channel-major [ch][s] (pre-transposed) -> all staging
// is aligned async b128, double buffered. One barrier per KV block; the P
// bf16 LDS round-trip is same-wave only (LDS ops in-order within a wave).
__global__ __launch_bounds__(256) void k_attn(
    const unsigned short* __restrict__ Qp, int qStride, int qHM, int qHA,
    const unsigned short* __restrict__ Kp, int kStride, int kHM, int kHA,
    const unsigned short* __restrict__ Vtg, int vStride,
    unsigned short* __restrict__ Op, int oStride,
    int Tq, int Skv, float scale2) {
  __shared__ unsigned short Kl[2][64 * 72];   // [buf][s][ch]
  __shared__ unsigned short Vl[2][64 * 72];   // [buf][ch][s]
  __shared__ unsigned short Pl[128 * 72];     // P staging, 16 rows per wave
  int t = threadIdx.x;
  int lane = t & 31, wv = t >> 5;
  int l16 = lane & 15, hi = lane >> 4;
  int b = blockIdx.x / Hh, h = blockIdx.x % Hh;
  int qRow0 = blockIdx.y * 128 + wv * 16;
  size_t qBase = (size_t)b * Tq, kBase = (size_t)b * Skv;
  size_t vRow = (size_t)(b * Hh + h) * CHd;
  int qOff = h * qHM + qHA, kOff = h * kHM + kHA;

  Frag qf[2];
  {
    const unsigned short* qr = Qp + (qBase + qRow0 + l16) * qStride + qOff;
#pragma unroll
    for (int kf = 0; kf < 2; kf++) {
      int c0 = kf * 32 + hi * 8;
      qf[kf].u[0] = *(const uint4*)(qr + c0);
      qf[kf].u[1] = *(const uint4*)(qr + c0 + 16);
    }
  }

  f32x8 oacc[4];
#pragma unroll
  for (int ni = 0; ni < 4; ni++) oacc[ni] = zero8();
  float mrow[8], lrow[8];
#pragma unroll
  for (int r = 0; r < 8; r++) { mrow[r] = -1e30f; lrow[r] = 0.f; }

  auto stageKV = [&](int sb, int buf) {
    int idx = t;
#pragma unroll
    for (int pp = 0; pp < 2; pp++, idx += 256) {
      int rr = idx >> 3, c0 = (idx & 7) * 8;
      stage16(&Kl[buf][rr * 72 + c0], Kp + (kBase + sb + rr) * kStride + kOff + c0);
      stage16(&Vl[buf][rr * 72 + c0], Vtg + (vRow + rr) * vStride + sb + c0);
    }
  };

  stageKV(0, 0);
  for (int sb = 0; sb < Skv; sb += 64) {
    int buf = (sb >> 6) & 1;
    wait_async0();
    __syncthreads();
    if (sb + 64 < Skv) stageKV(sb + 64, buf ^ 1);  // DMA overlaps compute

    // S = Q @ K^T  (16 x 64 per wave)
    f32x8 sacc[4];
#pragma unroll
    for (int ni = 0; ni < 4; ni++) sacc[ni] = zero8();
#pragma unroll
    for (int kf = 0; kf < 2; kf++) {
      int c0 = kf * 32 + hi * 8;
#pragma unroll
      for (int ni = 0; ni < 4; ni++) {
        Frag kfr;
        const unsigned short* p = &Kl[buf][(ni * 16 + l16) * 72];
        kfr.u[0] = *(const uint4*)(p + c0);
        kfr.u[1] = *(const uint4*)(p + c0 + 16);
        sacc[ni] = __builtin_amdgcn_wmma_f32_16x16x32_bf16(
            false, qf[kf].v, false, kfr.v, (short)0, sacc[ni], false, false);
      }
    }

    // online softmax (D layout: row = r + 8*hi, col = l16 within n-tile)
    float mnew[8], alpha[8];
#pragma unroll
    for (int r = 0; r < 8; r++) {
      float mx = -1e30f;
#pragma unroll
      for (int ni = 0; ni < 4; ni++) mx = fmaxf(mx, sacc[ni][r] * scale2);
#pragma unroll
      for (int msk = 8; msk >= 1; msk >>= 1) mx = fmaxf(mx, __shfl_xor(mx, msk, 32));
      mnew[r]  = fmaxf(mrow[r], mx);
      alpha[r] = __expf(mrow[r] - mnew[r]);
      mrow[r]  = mnew[r];
    }
#pragma unroll
    for (int r = 0; r < 8; r++) {
      float rsum = 0.f;
#pragma unroll
      for (int ni = 0; ni < 4; ni++) {
        float pv = __expf(sacc[ni][r] * scale2 - mnew[r]);
        rsum += pv;
        Pl[(wv * 16 + r + hi * 8) * 72 + ni * 16 + l16] = f2bf(pv);
      }
#pragma unroll
      for (int msk = 8; msk >= 1; msk >>= 1) rsum += __shfl_xor(rsum, msk, 32);
      lrow[r] = lrow[r] * alpha[r] + rsum;
#pragma unroll
      for (int ni = 0; ni < 4; ni++) oacc[ni][r] *= alpha[r];
    }

    // O += P @ V  (same-wave P round-trip; LDS ops in-order within wave)
#pragma unroll
    for (int kf = 0; kf < 2; kf++) {
      int c0 = kf * 32 + hi * 8;
      Frag pf;
      const unsigned short* pp = &Pl[(wv * 16 + l16) * 72];
      pf.u[0] = *(const uint4*)(pp + c0);
      pf.u[1] = *(const uint4*)(pp + c0 + 16);
#pragma unroll
      for (int ni = 0; ni < 4; ni++) {
        Frag vf;
        const unsigned short* vp = &Vl[buf][(ni * 16 + l16) * 72];
        vf.u[0] = *(const uint4*)(vp + c0);
        vf.u[1] = *(const uint4*)(vp + c0 + 16);
        oacc[ni] = __builtin_amdgcn_wmma_f32_16x16x32_bf16(
            false, pf.v, false, vf.v, (short)0, oacc[ni], false, false);
      }
    }
  }

#pragma unroll
  for (int r = 0; r < 8; r++) {
    float inv = 1.f / lrow[r];
    int row = qRow0 + r + hi * 8;
    unsigned short* orow = Op + (qBase + row) * (size_t)oStride + h * CHd;
#pragma unroll
    for (int ni = 0; ni < 4; ni++) orow[ni * 16 + l16] = f2bf(oacc[ni][r] * inv);
  }
}

// ------------------------------- launcher -----------------------------------
extern "C" void kernel_launch(void* const* d_in, const int* in_sizes, int n_in,
                              void* d_out, int out_size, void* d_ws, size_t ws_size,
                              hipStream_t stream) {
  const float* x      = (const float*)d_in[0];
  const float* cond_k = (const float*)d_in[1];
  const float* cond_v = (const float*)d_in[2];
  const float* qkv_w  = (const float*)d_in[3];
  const float* qkv_b  = (const float*)d_in[4];
  const float* apw    = (const float*)d_in[5];
  const float* apb    = (const float*)d_in[6];
  const float* cpw    = (const float*)d_in[7];
  const float* cpb    = (const float*)d_in[8];
  const float* fcw    = (const float*)d_in[9];
  const float* fcb    = (const float*)d_in[10];
  const float* mpw    = (const float*)d_in[11];
  const float* mpb    = (const float*)d_in[12];
  const float* ln1g   = (const float*)d_in[13];
  const float* ln1b   = (const float*)d_in[14];
  const float* ln3g   = (const float*)d_in[15];
  const float* ln3b   = (const float*)d_in[16];
  const float* ln4g   = (const float*)d_in[17];
  const float* ln4b   = (const float*)d_in[18];

  float* xo = (float*)d_out;  // running activation x lives in d_out (fp32)

  char* p = (char*)d_ws;
  auto carve = [&](size_t elems) {
    unsigned short* r = (unsigned short*)p;
    p += ((elems * 2 + 255) & ~(size_t)255);
    return r;
  };
  unsigned short* wq  = carve((size_t)LAY * Wd * 3 * Wd);  // qkv_w^T  (N,K) bf16
  unsigned short* wa  = carve((size_t)LAY * Wd * Wd);      // attn_proj^T
  unsigned short* wc  = carve((size_t)LAY * Wd * Wd);      // cross_proj^T
  unsigned short* wf  = carve((size_t)LAY * Wd * 4 * Wd);  // fc^T
  unsigned short* wmp = carve((size_t)LAY * 4 * Wd * Wd);  // mlp_proj^T
  unsigned short* kkb = carve((size_t)Bsz * Ssz * Wd);     // cond_k bf16 (row-major)
  unsigned short* vtb = carve((size_t)Bsz * Hh * CHd * Ssz); // cond_v (B,H,CH,S)
  unsigned short* hb  = carve((size_t)BT * Wd);            // LN output
  unsigned short* qb  = carve((size_t)BT * 3 * Wd);        // qkv (q,k used)
  unsigned short* vts = carve((size_t)Bsz * Hh * CHd * Tsz); // self V (B,H,CH,T)
  unsigned short* ab  = carve((size_t)BT * Wd);            // attn output
  unsigned short* mb  = carve((size_t)BT * 4 * Wd);        // MLP hidden

  hipMemcpyAsync(xo, x, (size_t)BT * Wd * sizeof(float), hipMemcpyDeviceToDevice, stream);

  auto grd = [](size_t n) { int b = (int)((n + 255) / 256); return b > 2048 ? 2048 : b; };
  size_t nq = (size_t)LAY * Wd * 3 * Wd, na = (size_t)LAY * Wd * Wd;
  size_t nf = (size_t)LAY * Wd * 4 * Wd;
  k_cvt_wt<<<grd(nq), 256, 0, stream>>>(qkv_w, wq, Wd, 3 * Wd, (int)nq);
  k_cvt_wt<<<grd(na), 256, 0, stream>>>(apw, wa, Wd, Wd, (int)na);
  k_cvt_wt<<<grd(na), 256, 0, stream>>>(cpw, wc, Wd, Wd, (int)na);
  k_cvt_wt<<<grd(nf), 256, 0, stream>>>(fcw, wf, Wd, 4 * Wd, (int)nf);
  k_cvt_wt<<<grd(nf), 256, 0, stream>>>(mpw, wmp, 4 * Wd, Wd, (int)nf);
  size_t nc = (size_t)Bsz * Ssz * Wd;
  k_cvt<<<grd(nc), 256, 0, stream>>>(cond_k, kkb, (int)nc);
  k_cvt_vt<<<grd(nc), 256, 0, stream>>>(cond_v, vtb, (int)nc);

  const float scale2 = 0.125f;  // (1/ch^0.25)^2 = 1/sqrt(64)

  for (int i = 0; i < LAY; i++) {
    k_ln<<<BT, 256, 0, stream>>>(xo, ln1g + i * Wd, ln1b + i * Wd, hb);
    // qkv = h @ qkv_w + b ; v-part written transposed to vts
    k_gemm<3><<<dim3(3 * Wd / 128, BT / 128), 256, 0, stream>>>(
        hb, Wd, wq + (size_t)i * Wd * 3 * Wd, Wd, qkv_b + (size_t)i * 3 * Wd,
        nullptr, qb, vts, BT, 3 * Wd, Wd);
    // self-attention
    k_attn<<<dim3(Bsz * Hh, Tsz / 128), 256, 0, stream>>>(
        qb, 3 * Wd, 3 * CHd, 0,
        qb, 3 * Wd, 3 * CHd, CHd,
        vts, Tsz,
        ab, Wd, Tsz, Tsz, scale2);
    // x += a @ attn_proj_w + b
    k_gemm<1><<<dim3(Wd / 128, BT / 128), 256, 0, stream>>>(
        ab, Wd, wa + (size_t)i * Wd * Wd, Wd, apb + (size_t)i * Wd, xo, xo, nullptr,
        BT, Wd, Wd);
    k_ln<<<BT, 256, 0, stream>>>(xo, ln3g + i * Wd, ln3b + i * Wd, hb);
    // cross-attention: Q = h, K/V = cond
    k_attn<<<dim3(Bsz * Hh, Tsz / 128), 256, 0, stream>>>(
        hb, Wd, CHd, 0,
        kkb, Wd, CHd, 0,
        vtb, Ssz,
        ab, Wd, Tsz, Ssz, scale2);
    // x += a @ cross_proj_w + b
    k_gemm<1><<<dim3(Wd / 128, BT / 128), 256, 0, stream>>>(
        ab, Wd, wc + (size_t)i * Wd * Wd, Wd, cpb + (size_t)i * Wd, xo, xo, nullptr,
        BT, Wd, Wd);
    k_ln<<<BT, 256, 0, stream>>>(xo, ln4g + i * Wd, ln4b + i * Wd, hb);
    // m = gelu(h @ fc_w + b)
    k_gemm<2><<<dim3(4 * Wd / 128, BT / 128), 256, 0, stream>>>(
        hb, Wd, wf + (size_t)i * Wd * 4 * Wd, Wd, fcb + (size_t)i * 4 * Wd,
        nullptr, mb, nullptr, BT, 4 * Wd, Wd);
    // x += m @ mlp_proj_w + b
    k_gemm<1><<<dim3(Wd / 128, BT / 128), 256, 0, stream>>>(
        mb, 4 * Wd, wmp + (size_t)i * 4 * Wd * Wd, 4 * Wd, mpb + (size_t)i * Wd, xo, xo,
        nullptr, BT, Wd, 4 * Wd);
  }
}
